// Attention_39548058861624
// MI455X (gfx1250) — matrix operand order repaired and checked by reading, not compiled, not run on hardware
//
#include <hip/hip_runtime.h>
#include <math.h>

// ---------------------------------------------------------------------------
// CDNA5 (gfx1250) attention pipeline, bf16 WMMA with f32 accumulation.
// Problem: B=2, T=2048, C=2048, H=16, hd=128.
// Workspace layout (assumes ws_size >= 112 MiB):
//   [  0MiB) xb   : x in bf16            [B*T, C]      16 MiB
//   [ 16MiB) wqb  : Wq bf16              [C, C]         8 MiB
//   [ 24MiB) wkb  : Wk bf16                             8 MiB
//   [ 32MiB) wvb  : Wv bf16                             8 MiB
//   [ 40MiB) wob  : Wo bf16                             8 MiB
//   [ 48MiB) Qb   : Q bf16 (RoPE'd)      [B,H,T,hd]    16 MiB
//   [ 64MiB) Kb   : K bf16 (RoPE'd)      [B,H,T,hd]    16 MiB
//   [ 80MiB) Vb   : V bf16               [B,H,T,hd]    16 MiB
//   [ 96MiB) Ab   : attn out bf16        [B*T, C]      16 MiB
// ---------------------------------------------------------------------------

typedef __bf16 bf16_t;
typedef __attribute__((ext_vector_type(16))) __bf16 v16bf;
typedef __attribute__((ext_vector_type(8)))  float  v8f;
typedef unsigned short u16;
typedef __attribute__((ext_vector_type(8))) unsigned short v8u;
typedef __attribute__((ext_vector_type(4))) unsigned int u32x4;
typedef __attribute__((ext_vector_type(8))) int i32x8;
typedef __attribute__((ext_vector_type(4))) int i32x4;

union Frag {
    v16bf bf;
    v8u   h[2];
};

__device__ __forceinline__ u16 f32_to_bf16(float f) {
    unsigned u = __float_as_uint(f);
    unsigned r = (u + 0x7FFFu + ((u >> 16) & 1u)) >> 16;
    return (u16)r;
}
__device__ __forceinline__ float bf16_to_f32(u16 h) {
    return __uint_as_float(((unsigned)h) << 16);
}

// Tensor Data Mover availability (guarded: fallback = cooperative loads)
#if defined(__has_builtin)
#if __has_builtin(__builtin_amdgcn_tensor_load_to_lds) && __has_builtin(__builtin_amdgcn_s_wait_tensorcnt)
#define USE_TDM 1
#endif
#endif
#ifndef USE_TDM
#define USE_TDM 0
#endif

// ---------------------------------------------------------------------------
// 1) f32 -> bf16 conversion (grid-stride)
// ---------------------------------------------------------------------------
__global__ void cvt_f32_bf16(const float* __restrict__ in, u16* __restrict__ out, int n) {
    int i = blockIdx.x * blockDim.x + threadIdx.x;
    int stride = gridDim.x * blockDim.x;
    for (; i < n; i += stride) out[i] = f32_to_bf16(in[i]);
}

// ---------------------------------------------------------------------------
// 2) bf16 GEMM: C[m,n] = sum_k A[m,k] * W[n,k]   (y = x @ W^T, torch Linear)
//    Tile 128x128, BK=32, 256 threads (8 waves, wave grid 2x4, 64x32/wave).
//    Triple-buffered LDS; tiles staged by the Tensor Data Mover when
//    available (wave 0 issues two stages ahead, TENSORcnt<=2 gated),
//    else cooperative b128 loads two stages ahead.
//    mode 0: bf16 out remapped to [B,H,T,hd]    (QKV projections)
//    mode 2: f32 out flat [m*N+n]               (final Wo projection)
// ---------------------------------------------------------------------------
#define GBM 128
#define GBN 128
#define GBK 32
#define GLD 40   // padded LDS row stride (elements); 80B rows, stride%64dw == 20
                 // -> fragment rows hit 16 distinct banks (conflict-free)

#if USE_TDM
// One 2D bf16 tile: GBM rows x GBK cols, row stride = tensorCols elements.
// TDM pad feature: +4 dwords after every 16 dwords -> LDS row stride 80B (= GLD).
__device__ __forceinline__ void tdm_load_tile(
    const u16* gtile, unsigned ldsOff, int tensorRows, int tensorCols)
{
    unsigned long long ga = (unsigned long long)(size_t)gtile;
    u32x4 g0;
    g0[0] = 1u;                                             // count=1, user mode
    g0[1] = ldsOff;                                         // lds_addr (bytes)
    g0[2] = (unsigned)(ga & 0xffffffffu);                   // global_addr lo
    g0[3] = (unsigned)((ga >> 32) & 0x1ffffffu) | (2u << 30); // addr hi | type=2
    i32x8 g1;
    g1[0] = (int)((1u << 16)      // data_size = 2 bytes
                | (1u << 20)      // pad_enable
                | (3u << 22)      // pad_interval: 16 dwords (= one 64B row)
                | (3u << 25));    // pad_amount:   4 dwords (16B) -> 80B stride
    g1[1] = (int)(((unsigned)tensorCols & 0xffffu) << 16);              // dim0 lo16
    g1[2] = (int)((((unsigned)tensorCols >> 16) & 0xffffu)
                | (((unsigned)tensorRows & 0xffffu) << 16));            // dim0 hi/dim1 lo
    g1[3] = (int)((((unsigned)tensorRows >> 16) & 0xffffu)
                | ((unsigned)GBK << 16));                               // dim1 hi/tile_dim0
    g1[4] = (int)GBM;                                                   // tile_dim1 (tile_dim2=0)
    g1[5] = (int)(unsigned)tensorCols;                                  // dim0_stride lo32
    g1[6] = 0;
    g1[7] = 0;
    i32x4 gz = {0, 0, 0, 0};
#if __clang_major__ >= 23
    i32x8 gz8 = {0, 0, 0, 0, 0, 0, 0, 0};
    __builtin_amdgcn_tensor_load_to_lds(g0, g1, gz, gz, gz8, 0);
#else
    __builtin_amdgcn_tensor_load_to_lds(g0, g1, gz, gz, 0);
#endif
}
#endif

__global__ __launch_bounds__(256) void gemm_bf16(
    const u16* __restrict__ A, const u16* __restrict__ W,
    void* __restrict__ Out, int M, int N, int K, int mode)
{
    __shared__ u16 As[3][GBM][GLD];   // 30 KiB
    __shared__ u16 Bs[3][GBN][GLD];   // 30 KiB

    const int tid   = threadIdx.x;
    const int wave  = tid >> 5;
    const int lane  = tid & 31;
    const int l16   = lane & 15;
    const int halfA = (lane < 16) ? 0 : 8;   // A-operand K sub-offset
    const int halfB = (lane < 16) ? 0 : 16;  // B-operand K sub-offset
    const int rAdd  = (lane < 16) ? 0 : 8;   // C-layout row offset
    const int waveM = wave >> 2;             // 0..1  (64-row slab)
    const int waveN = wave & 3;              // 0..3  (32-col slab)

    const int rowBase = blockIdx.y * GBM;
    const int colBase = blockIdx.x * GBN;

    v8f zero = {};
    v8f acc[4][2];
    for (int i = 0; i < 4; i++)
        for (int j = 0; j < 2; j++) acc[i][j] = zero;

    const int ldr = tid >> 1;          // 0..127: tile row to load (fallback path)
    const int ldc = (tid & 1) * 16;    // 0 / 16: 16-element half-chunk
    const u16* Arow = A + (size_t)(rowBase + ldr) * K;
    const u16* Wrow = W + (size_t)(colBase + ldr) * K;
    (void)Arow; (void)Wrow;

    // ---- tile stage: TDM (wave 0 issues DMA) or cooperative loads ----
#if USE_TDM
#define ISSUE_TILE(buf_, k0_)                                                   \
    do {                                                                        \
        if (wave == 0) {                                                        \
            tdm_load_tile(A + (size_t)rowBase * K + (k0_),                      \
                          (unsigned)(size_t)&As[(buf_)][0][0], M, K);           \
            tdm_load_tile(W + (size_t)colBase * K + (k0_),                      \
                          (unsigned)(size_t)&Bs[(buf_)][0][0], N, K);           \
        }                                                                       \
    } while (0)
#else
#define ISSUE_TILE(buf_, k0_)                                                   \
    do {                                                                        \
        *(v8u*)&As[(buf_)][ldr][ldc] = *(const v8u*)(Arow + (k0_) + ldc);       \
        *(v8u*)&Bs[(buf_)][ldr][ldc] = *(const v8u*)(Wrow + (k0_) + ldc);       \
    } while (0)
#endif

    ISSUE_TILE(0, 0);
    ISSUE_TILE(1, GBK);
    int buf = 0;
    for (int k0 = 0; k0 < K; k0 += GBK) {
#if USE_TDM
        // 2 descriptors/stage, in-order completion: <=2 outstanding means the
        // current stage has landed (unless it is the last one in flight).
        if (wave == 0) {
            if (k0 + GBK < K) __builtin_amdgcn_s_wait_tensorcnt(2);
            else              __builtin_amdgcn_s_wait_tensorcnt(0);
        }
#endif
        __syncthreads();                       // current buf's tile is resident
        if (k0 + 2 * GBK < K) {                // keep the mover 2 stages ahead
            int nb = buf + 2; if (nb >= 3) nb -= 3;
            ISSUE_TILE(nb, k0 + 2 * GBK);
        }

        Frag a[4], b[2];
        for (int i = 0; i < 4; i++) {
            int r = waveM * 64 + i * 16 + l16;
            a[i].h[0] = *(const v8u*)&As[buf][r][halfA];        // K 0..7  (8..15)
            a[i].h[1] = *(const v8u*)&As[buf][r][16 + halfA];   // K 16..23 (24..31)
        }
        for (int j = 0; j < 2; j++) {
            int c = waveN * 32 + j * 16 + l16;
            b[j].h[0] = *(const v8u*)&Bs[buf][c][halfA];        // K 0..7  (8..15)
            b[j].h[1] = *(const v8u*)&Bs[buf][c][16 + halfA];   // K 16..23 (24..31)
        }

        for (int i = 0; i < 4; i++)
            for (int j = 0; j < 2; j++)
                acc[i][j] = __builtin_amdgcn_wmma_f32_16x16x32_bf16(
                    false, a[i].bf, false, b[j].bf, (short)0, acc[i][j], false, false);
        __syncthreads();
        buf += 1; if (buf >= 3) buf = 0;
    }
#undef ISSUE_TILE

    // epilogue: C layout -> VGPR r holds (M=r | r+8, N=lane%16)
    for (int i = 0; i < 4; i++) {
        for (int j = 0; j < 2; j++) {
            int n = colBase + waveN * 32 + j * 16 + l16;
            for (int r = 0; r < 8; r++) {
                int m = rowBase + waveM * 64 + i * 16 + rAdd + r;
                float v = acc[i][j][r];
                if (mode == 2) {
                    ((float*)Out)[(size_t)m * N + n] = v;
                } else {
                    // m = b*T + t ; n = h*hd + d  ->  [B,H,T,hd]
                    int bb = m >> 11, t = m & 2047;
                    int h  = n >> 7,  d = n & 127;
                    size_t idx = ((((size_t)bb * 16 + h) * 2048) + t) * 128 + d;
                    ((u16*)Out)[idx] = f32_to_bf16(v);
                }
            }
        }
    }
}

// ---------------------------------------------------------------------------
// 3) RoPE on Q and K in-place ([B,H,T,128] bf16). One thread per (b,h,t,d<64)
//    pair: q'[d] = q[d]*cos - q[d+64]*sin ; q'[d+64] = q[d+64]*cos + q[d]*sin
//    10000^(-2d/128) = exp2(-d * log2(10000)/64): single v_exp_f32.
// ---------------------------------------------------------------------------
__global__ void rope_kernel(u16* __restrict__ Q, u16* __restrict__ Kt, int total) {
    int i = blockIdx.x * blockDim.x + threadIdx.x;
    if (i >= total) return;
    int d = i & 63;
    int rest = i >> 6;              // (b*H + h)*T + t
    int t = rest & 2047;
    size_t base = (size_t)rest * 128;

    const float kLog2BaseOver64 = 0.2076205059433286f;   // log2(10000)/64
    float inv = exp2f(-(float)d * kLog2BaseOver64);
    float ang = (float)t * inv;
    float c = __cosf(ang), s = __sinf(ang);

    {
        float x1 = bf16_to_f32(Q[base + d]);
        float x2 = bf16_to_f32(Q[base + d + 64]);
        Q[base + d]      = f32_to_bf16(x1 * c - x2 * s);
        Q[base + d + 64] = f32_to_bf16(x2 * c + x1 * s);
    }
    {
        float x1 = bf16_to_f32(Kt[base + d]);
        float x2 = bf16_to_f32(Kt[base + d + 64]);
        Kt[base + d]      = f32_to_bf16(x1 * c - x2 * s);
        Kt[base + d + 64] = f32_to_bf16(x2 * c + x1 * s);
    }
}

// ---------------------------------------------------------------------------
// 4) Causal flash attention. Block = 256 threads (8 waves); 128 queries/block
//    (16 per wave); KV tiles of 64 staged in LDS (K row-major, V transposed).
//    Online softmax with 16-lane shuffle reductions (matches WMMA C layout).
//    All LDS row strides chosen conflict-free: stride%64dw in {4,20,36}.
// ---------------------------------------------------------------------------
#define FBQ  128
#define FBKV 64
#define FHD  128
#define FT   2048
#define KLD  (FHD + 8)    // 136 el = 272B rows: stride%64dw == 4  -> conflict-free
#define VLD  (FBKV + 8)   // 72 el = 144B rows:  stride%64dw == 36 -> conflict-free
#define PLD  (FBKV + 8)   // 72 el rows for P

__global__ __launch_bounds__(256) void flash_kernel(
    const u16* __restrict__ Q, const u16* __restrict__ Kg,
    const u16* __restrict__ Vg, u16* __restrict__ Outb)
{
    __shared__ u16 Ks[FBKV][KLD];          // 17 KiB : K tile, row-major [kv][d]
    __shared__ u16 Vt[FHD][VLD];           // 18 KiB : V tile transposed [d][kv]
    __shared__ u16 Ps[8][16][PLD];         // 18 KiB : per-wave P (C->A relayout)

    const int tid   = threadIdx.x;
    const int wave  = tid >> 5;
    const int lane  = tid & 31;
    const int l16   = lane & 15;
    const int halfA = (lane < 16) ? 0 : 8;
    const int halfB = (lane < 16) ? 0 : 16;
    const int rAdd  = (lane < 16) ? 0 : 8;

    const int qBase = blockIdx.x * FBQ;
    const int bh    = blockIdx.y;          // b*H + h
    const size_t headOff = (size_t)bh * FT * FHD;

    // Q fragments for this wave's 16 rows, K-dim = 128 -> 4 fragments, kept live
    Frag qa[4];
    {
        const u16* qrow = Q + headOff + (size_t)(qBase + wave * 16 + l16) * FHD;
        for (int kk = 0; kk < 4; kk++) {
            qa[kk].h[0] = *(const v8u*)(qrow + kk * 32 + halfA);
            qa[kk].h[1] = *(const v8u*)(qrow + kk * 32 + 16 + halfA);
        }
    }

    v8f zero = {};
    v8f o[8];
    for (int i = 0; i < 8; i++) o[i] = zero;
    float mrow[8], lrow[8];
    for (int r = 0; r < 8; r++) { mrow[r] = -1e30f; lrow[r] = 0.0f; }

    const float scale = 0.08838834764831845f;   // 1/sqrt(128)
    const int jmax = (qBase / FBKV) + 2;        // causal: tiles with kv < qBase+128

    const int str = tid >> 2;               // 0..63  (stage row)
    const int stc = (tid & 3) * 32;         // 0,32,64,96 (stage col)

    for (int j = 0; j < jmax; j++) {
        const int kvBase = j * FBKV;

        // cooperative stage: K row-major, V transposed (for B-operand reads)
        {
            const v8u* ksrc = (const v8u*)(Kg + headOff + (size_t)(kvBase + str) * FHD + stc);
            *(v8u*)&Ks[str][stc]     = ksrc[0];
            *(v8u*)&Ks[str][stc + 8] = ksrc[1];
            const u16* vsrc = Vg + headOff + (size_t)(kvBase + str) * FHD + stc;
            for (int e = 0; e < 32; e++) Vt[stc + e][str] = vsrc[e];
        }
        // prefetch next tile while this one is consumed
        if (j + 1 < jmax) {
            __builtin_prefetch(Kg + headOff + (size_t)(kvBase + FBKV + str) * FHD + stc, 0, 3);
            __builtin_prefetch(Vg + headOff + (size_t)(kvBase + FBKV + str) * FHD + stc, 0, 3);
        }
        __syncthreads();

        // S = scale * (Q K^T) for 16 rows x 64 cols
        v8f s[4];
        for (int nt = 0; nt < 4; nt++) {
            s[nt] = zero;
            for (int kk = 0; kk < 4; kk++) {
                Frag kb;
                int c = nt * 16 + l16;
                kb.h[0] = *(const v8u*)&Ks[c][kk * 32 + halfA];
                kb.h[1] = *(const v8u*)&Ks[c][kk * 32 + 16 + halfA];
                s[nt] = __builtin_amdgcn_wmma_f32_16x16x32_bf16(
                    false, qa[kk].bf, false, kb.bf, (short)0, s[nt], false, false);
            }
        }

        // scale + causal mask
        const int qRow0 = qBase + wave * 16 + rAdd;
        for (int nt = 0; nt < 4; nt++) {
            int col = kvBase + nt * 16 + l16;
            for (int r = 0; r < 8; r++) {
                float v = s[nt][r] * scale;
                if (col > qRow0 + r) v = -1e9f;
                s[nt][r] = v;
            }
        }

        // online softmax: row max (16-lane reduce), rescale, exp, row sum
        float mnew[8], alpha[8];
        for (int r = 0; r < 8; r++) {
            float v = fmaxf(fmaxf(s[0][r], s[1][r]), fmaxf(s[2][r], s[3][r]));
            for (int off = 8; off >= 1; off >>= 1)
                v = fmaxf(v, __shfl_xor(v, off, 32));
            mnew[r]  = fmaxf(mrow[r], v);
            alpha[r] = __expf(mrow[r] - mnew[r]);
            mrow[r]  = mnew[r];
        }
        float rsum[8];
        for (int r = 0; r < 8; r++) rsum[r] = 0.0f;
        for (int nt = 0; nt < 4; nt++) {
            for (int r = 0; r < 8; r++) {
                float p = __expf(s[nt][r] - mnew[r]);
                s[nt][r] = p;
                rsum[r] += p;
            }
        }
        for (int r = 0; r < 8; r++) {
            float v = rsum[r];
            for (int off = 8; off >= 1; off >>= 1) v += __shfl_xor(v, off, 32);
            lrow[r] = lrow[r] * alpha[r] + v;
        }

        // relayout P (C-layout regs) -> wave-private LDS row-major bf16.
        // LDS ops from the same wave complete in order; no barrier needed.
        for (int nt = 0; nt < 4; nt++)
            for (int r = 0; r < 8; r++)
                Ps[wave][rAdd + r][nt * 16 + l16] = f32_to_bf16(s[nt][r]);

        Frag pa[2];
        {
            const u16* prow = &Ps[wave][l16][0];
            for (int kk = 0; kk < 2; kk++) {
                pa[kk].h[0] = *(const v8u*)(prow + kk * 32 + halfA);
                pa[kk].h[1] = *(const v8u*)(prow + kk * 32 + 16 + halfA);
            }
        }

        // O = O*alpha + P @ V
        for (int dt = 0; dt < 8; dt++) {
            for (int r = 0; r < 8; r++) o[dt][r] *= alpha[r];
            for (int kk = 0; kk < 2; kk++) {
                Frag vb;
                int d = dt * 16 + l16;
                vb.h[0] = *(const v8u*)&Vt[d][kk * 32 + halfA];
                vb.h[1] = *(const v8u*)&Vt[d][kk * 32 + 16 + halfA];
                o[dt] = __builtin_amdgcn_wmma_f32_16x16x32_bf16(
                    false, pa[kk].bf, false, vb.bf, (short)0, o[dt], false, false);
            }
        }
        __syncthreads();
    }

    // normalize + store bf16 to [B*T, C]:  row = b*T + q, col = h*128 + d
    const int b = bh >> 4;
    const int h = bh & 15;
    for (int dt = 0; dt < 8; dt++) {
        int d = dt * 16 + l16;
        for (int r = 0; r < 8; r++) {
            int qrow = qBase + wave * 16 + rAdd + r;
            float v = o[dt][r] / lrow[r];
            Outb[((size_t)b * FT + qrow) * 2048 + h * FHD + d] = f32_to_bf16(v);
        }
    }
}

// ---------------------------------------------------------------------------
// Host orchestration
// ---------------------------------------------------------------------------
extern "C" void kernel_launch(void* const* d_in, const int* in_sizes, int n_in,
                              void* d_out, int out_size, void* d_ws, size_t ws_size,
                              hipStream_t stream) {
    const float* x  = (const float*)d_in[0];
    // d_in[1] = mask (recomputed in-kernel, unused)
    const float* Wq = (const float*)d_in[2];
    const float* Wk = (const float*)d_in[3];
    const float* Wv = (const float*)d_in[4];
    const float* Wo = (const float*)d_in[5];

    const int B = 2, T = 2048, C = 2048, H = 16;
    const size_t MB = 1024 * 1024;
    char* ws = (char*)d_ws;
    u16* xb  = (u16*)(ws + 0 * MB);
    u16* wqb = (u16*)(ws + 16 * MB);
    u16* wkb = (u16*)(ws + 24 * MB);
    u16* wvb = (u16*)(ws + 32 * MB);
    u16* wob = (u16*)(ws + 40 * MB);
    u16* Qb  = (u16*)(ws + 48 * MB);
    u16* Kb  = (u16*)(ws + 64 * MB);
    u16* Vb  = (u16*)(ws + 80 * MB);
    u16* Ab  = (u16*)(ws + 96 * MB);

    const int nx = B * T * C;   // 8,388,608
    const int nw = C * C;       // 4,194,304

    cvt_f32_bf16<<<4096, 256, 0, stream>>>(x,  xb,  nx);
    cvt_f32_bf16<<<2048, 256, 0, stream>>>(Wq, wqb, nw);
    cvt_f32_bf16<<<2048, 256, 0, stream>>>(Wk, wkb, nw);
    cvt_f32_bf16<<<2048, 256, 0, stream>>>(Wv, wvb, nw);
    cvt_f32_bf16<<<2048, 256, 0, stream>>>(Wo, wob, nw);

    dim3 ggrid(C / GBN, (B * T) / GBM);   // (16, 32)
    gemm_bf16<<<ggrid, 256, 0, stream>>>(xb, wqb, Qb, B * T, C, C, 0);
    gemm_bf16<<<ggrid, 256, 0, stream>>>(xb, wkb, Kb, B * T, C, C, 0);
    gemm_bf16<<<ggrid, 256, 0, stream>>>(xb, wvb, Vb, B * T, C, C, 0);

    const int nrope = B * H * T * 64;     // 4,194,304 pairs
    rope_kernel<<<nrope / 256, 256, 0, stream>>>(Qb, Kb, nrope);

    flash_kernel<<<dim3(T / FBQ, B * H), 256, 0, stream>>>(Qb, Kb, Vb, Ab);

    gemm_bf16<<<ggrid, 256, 0, stream>>>(Ab, wob, d_out, B * T, C, C, 2);
}
